// FFNwMoE_69252052680928
// MI455X (gfx1250) — compile-verified, matrix operand order checked by reading, not run on hardware
//
#include <hip/hip_runtime.h>
#include <hip/hip_bf16.h>

// ---------------------------------------------------------------------------
// MoE FFN (SwiGLU, 8 experts, dense like the reference) for MI455X / gfx1250.
//   - One memory-bound pre-pass converts x/w1/w3/w2 to bf16 (padded) in ws.
//   - Both GEMMs run v_wmma_f32_16x16x32_bf16 with ALL tile staging done by
//     the Tensor Data Mover (tensor_load_to_lds), double-buffered in LDS and
//     synchronized with s_wait_tensorcnt + workgroup barriers.
// Shapes: T=8192, D=1024, H=684 (padded to 704 = 22*32), E=8.
// ---------------------------------------------------------------------------

typedef __attribute__((ext_vector_type(16))) __bf16 v16bf;
typedef __attribute__((ext_vector_type(8)))  float  v8f;
typedef __attribute__((ext_vector_type(4)))  unsigned int v4u;
typedef __attribute__((ext_vector_type(8)))  int    v8i;
typedef __attribute__((ext_vector_type(4)))  int    v4i;

#define NUM_EXPERTS 8
#define TOKENS      8192
#define DIM         1024
#define HID         684
#define HID_PAD     704      // 22 * 32
#define KC          32       // K chunk per WMMA stage
#define TILE_M      128      // 4 waves of 32 rows
#define TILE_N      64       // 2 waves of 32 cols

#if defined(__has_builtin)
#  if __has_builtin(__builtin_amdgcn_tensor_load_to_lds) && \
      __has_builtin(__builtin_amdgcn_s_wait_tensorcnt)
#    define HAVE_TDM 1
#  endif
#endif

#if defined(HAVE_TDM)
#  define BUF(i) ((i) & 1)
#else
#  define BUF(i) 0
#endif

static __device__ inline unsigned short f2bf(float f) {
    unsigned u = __builtin_bit_cast(unsigned, f);
    u += 0x7FFFu + ((u >> 16) & 1u);           // round-to-nearest-even
    return (unsigned short)(u >> 16);
}
static __device__ inline unsigned pack2(float a, float b) {
    return (unsigned)f2bf(a) | ((unsigned)f2bf(b) << 16);
}
static __device__ inline unsigned lds_ofs(const void* p) {
    return (unsigned)(uintptr_t)p;             // low 32 bits = LDS byte offset
}

#if defined(HAVE_TDM)
// 2-D TDM tile load: tile_rows x KC bf16 elements, packed row-major into LDS.
// D# per ISA ch.8: group0 {count, lds_addr, global_addr, type=2},
// group1 {data_size=2B, tensor dims (remaining), tile dims, dim0 stride}.
static __device__ inline void tdm_load_2d(unsigned lds_off, const void* gptr,
                                          unsigned tile_rows, unsigned stride_elems,
                                          unsigned rem0, unsigned rem1) {
    unsigned long long ga = (unsigned long long)(uintptr_t)gptr;
    v4u g0;
    g0[0] = 1u;                                              // count=1
    g0[1] = lds_off;                                         // lds_addr
    g0[2] = (unsigned)ga;                                    // global_addr lo
    g0[3] = ((unsigned)(ga >> 32) & 0x01FFFFFFu) | 0x80000000u; // type=2
    v8i g1;
    g1[0] = (int)(1u << 16);                                 // data_size = 2B
    g1[1] = (int)((rem0 & 0xFFFFu) << 16);                   // tensor_dim0
    g1[2] = (int)((rem0 >> 16) | ((rem1 & 0xFFFFu) << 16));  // tensor_dim1
    g1[3] = (int)((rem1 >> 16) | ((unsigned)KC << 16));      // tile_dim0 = KC
    g1[4] = (int)tile_rows;                                  // tile_dim1
    g1[5] = (int)stride_elems;                               // dim0 stride
    g1[6] = 0;
    g1[7] = 0;
    v4i g2 = {0, 0, 0, 0};
    v4i g3 = {0, 0, 0, 0};
#if __clang_major__ >= 23
    v8i g4 = {0, 0, 0, 0, 0, 0, 0, 0};
    __builtin_amdgcn_tensor_load_to_lds(g0, g1, g2, g3, g4, 0);
#else
    __builtin_amdgcn_tensor_load_to_lds(g0, g1, g2, g3, 0);
#endif
}
#else
// Fallback: cooperative uint4 copy of a bf16 tile (rows x KC) into LDS.
static __device__ inline void copy_tile(unsigned short* dst, const unsigned short* src,
                                        int rows, int stride, int tid) {
    const int nvec = rows * KC / 8;            // uint4 = 8 bf16
    for (int i = tid; i < nvec; i += 256) {
        int r = i >> 2, q = i & 3;
        *(uint4*)&dst[r * KC + q * 8] =
            ((const uint4*)(src + (size_t)r * stride))[q];
    }
}
#endif

// Load a 16x32 bf16 A/B fragment from LDS (row stride = 32 elements).
// ISA layout (16-bit 16x32): lane<16 -> K = {0..7, 16..23}; lane>=16 -> +8.
static __device__ inline v16bf load_frag(const unsigned short* base, int row, int lane) {
    const unsigned short* p = base + row * KC + ((lane >> 4) << 3);
    v16bf v;
#pragma unroll
    for (int i = 0; i < 8; ++i) {
        v[i]     = __builtin_bit_cast(__bf16, p[i]);
        v[i + 8] = __builtin_bit_cast(__bf16, p[i + 16]);
    }
    return v;
}

static __device__ inline v8f wmma_bf16(v16bf a, v16bf b, v8f c) {
    return __builtin_amdgcn_wmma_f32_16x16x32_bf16(
        false, a, false, b, (short)0, c, false, false);
}

// ---------------------------------------------------------------------------
// Zero out[T*D] (float4) and the 16 aux accumulators.
// ---------------------------------------------------------------------------
__global__ __launch_bounds__(256) void moe_zero(float* __restrict__ out,
                                                float* __restrict__ aux) {
    size_t i = (size_t)blockIdx.x * blockDim.x + threadIdx.x;
    float4 z; z.x = 0.f; z.y = 0.f; z.z = 0.f; z.w = 0.f;
    ((float4*)out)[i] = z;
    if (blockIdx.x == 0 && threadIdx.x < 16) aux[threadIdx.x] = 0.f;
}

// ---------------------------------------------------------------------------
// Padded fp32 -> bf16 conversion: dst[e][r][c] = (r<src_rows && c<src_cols) ?
// bf16(src[e][r][c]) : 0.   grid.y = expert; one thread per 4 dst elements.
// ---------------------------------------------------------------------------
__global__ __launch_bounds__(256) void conv_bf16_pad(const float* __restrict__ src,
                                                     unsigned short* __restrict__ dst,
                                                     int src_rows, int src_cols,
                                                     int dst_cols,
                                                     long long src_estride,
                                                     long long dst_estride) {
    const float* s = src + (size_t)blockIdx.y * src_estride;
    unsigned short* d = dst + (size_t)blockIdx.y * dst_estride;
    size_t idx = (size_t)blockIdx.x * 256 + threadIdx.x;
    int c4n = dst_cols >> 2;
    int r = (int)(idx / c4n);
    int c = (int)(idx % c4n) * 4;
    float4 v;
    if (r < src_rows && c < src_cols)
        v = *(const float4*)(s + (size_t)r * src_cols + c);
    else { v.x = v.y = v.z = v.w = 0.f; }
    uint2 u; u.x = pack2(v.x, v.y); u.y = pack2(v.z, v.w);
    *(uint2*)(d + (size_t)r * dst_cols + c) = u;
}

// ---------------------------------------------------------------------------
// Router: softmax(x @ router_w^T), top-2 -> combine[T,8]; aux partial sums.
// ---------------------------------------------------------------------------
__global__ __launch_bounds__(256) void moe_router(const float* __restrict__ x,
                                                  const float* __restrict__ rw,
                                                  float* __restrict__ combine,
                                                  float* __restrict__ aux) {
    __shared__ float rws[NUM_EXPERTS * DIM];   // 32 KB
    __shared__ float la[16];
    const int tid = threadIdx.x;
    {
        const float4* r4 = (const float4*)rw;
        float4* s4 = (float4*)rws;
        for (int i = tid; i < NUM_EXPERTS * DIM / 4; i += 256) s4[i] = r4[i];
    }
    if (tid < 16) la[tid] = 0.f;
    __syncthreads();

    const int t = blockIdx.x * 256 + tid;
    const float4* xp4 = (const float4*)(x + (size_t)t * DIM);
    float acc[NUM_EXPERTS];
#pragma unroll
    for (int e = 0; e < NUM_EXPERTS; ++e) acc[e] = 0.f;
    for (int d4 = 0; d4 < DIM / 4; ++d4) {
        float4 xv = xp4[d4];
#pragma unroll
        for (int e = 0; e < NUM_EXPERTS; ++e) {
            float4 wv = ((const float4*)&rws[e * DIM])[d4];
            acc[e] += xv.x * wv.x + xv.y * wv.y + xv.z * wv.z + xv.w * wv.w;
        }
    }
    float m = acc[0];
#pragma unroll
    for (int e = 1; e < NUM_EXPERTS; ++e) m = fmaxf(m, acc[e]);
    float p[NUM_EXPERTS], s = 0.f;
#pragma unroll
    for (int e = 0; e < NUM_EXPERTS; ++e) { p[e] = __expf(acc[e] - m); s += p[e]; }
    float inv = 1.f / s;
#pragma unroll
    for (int e = 0; e < NUM_EXPERTS; ++e) p[e] *= inv;
    int i1 = 0; float p1 = p[0];
#pragma unroll
    for (int e = 1; e < NUM_EXPERTS; ++e) if (p[e] > p1) { p1 = p[e]; i1 = e; }
    int i2 = -1; float p2 = -1.f;
#pragma unroll
    for (int e = 0; e < NUM_EXPERTS; ++e)
        if (e != i1 && p[e] > p2) { p2 = p[e]; i2 = e; }
#pragma unroll
    for (int e = 0; e < NUM_EXPERTS; ++e) {
        float c = (e == i1) ? p1 : ((e == i2) ? p2 : 0.f);
        combine[(size_t)t * NUM_EXPERTS + e] = c;
    }
    atomicAdd(&la[i1], 1.0f);
#pragma unroll
    for (int e = 0; e < NUM_EXPERTS; ++e) atomicAdd(&la[8 + e], p[e]);
    __syncthreads();
    if (tid < 16) atomicAdd(&aux[tid], la[tid]);
}

// aux_loss = 0.01 * sum_e(density_e * mean_prob_e) * 8
__global__ void moe_aux(const float* __restrict__ aux, float* __restrict__ out_aux) {
    if (threadIdx.x == 0 && blockIdx.x == 0) {
        const float invT = 1.0f / (float)TOKENS;
        float s = 0.f;
#pragma unroll
        for (int e = 0; e < NUM_EXPERTS; ++e)
            s += (aux[e] * invT) * (aux[8 + e] * invT);
        *out_aux = 0.01f * s * (float)NUM_EXPERTS;
    }
}

// ---------------------------------------------------------------------------
// FFN pass 1: h = silu(x @ w1^T) * (x @ w3^T) -> bf16 [T, HID_PAD].
// All operands bf16; tiles staged by TDM, double-buffered.
// grid = (T/128, HID_PAD/64), block = 256 (8 wave32s, 4x2 of 32x32 tiles).
// ---------------------------------------------------------------------------
__global__ __launch_bounds__(256) void moe_ffn1(const unsigned short* __restrict__ xbf,
                                                const unsigned short* __restrict__ w1p,
                                                const unsigned short* __restrict__ w3p,
                                                unsigned short* __restrict__ hbuf) {
    __shared__ unsigned short xs [2][TILE_M * KC];   // 2 x 8 KB
    __shared__ unsigned short w1s[2][TILE_N * KC];   // 2 x 4 KB
    __shared__ unsigned short w3s[2][TILE_N * KC];   // 2 x 4 KB

    const int tid  = threadIdx.x;
    const int lane = tid & 31;
    const int wid  = tid >> 5;
    const int wm   = wid & 3;
    const int wn   = wid >> 2;
    const int m0   = blockIdx.x * TILE_M;
    const int n0   = blockIdx.y * TILE_N;

    v8f zf = {0.f, 0.f, 0.f, 0.f, 0.f, 0.f, 0.f, 0.f};
    v8f acc1[2][2] = {{zf, zf}, {zf, zf}};
    v8f acc3[2][2] = {{zf, zf}, {zf, zf}};

    const int nIter = DIM / KC;
#if defined(HAVE_TDM)
    // prologue: DMA tile 0
    if (wid == 0) tdm_load_2d(lds_ofs(&xs[0][0]),  xbf + (size_t)m0 * DIM, TILE_M, DIM, DIM, TOKENS - m0);
    if (wid == 1) tdm_load_2d(lds_ofs(&w1s[0][0]), w1p + (size_t)n0 * DIM, TILE_N, DIM, DIM, HID_PAD - n0);
    if (wid == 2) tdm_load_2d(lds_ofs(&w3s[0][0]), w3p + (size_t)n0 * DIM, TILE_N, DIM, DIM, HID_PAD - n0);
#endif
    for (int i = 0; i < nIter; ++i) {
        const int cur = BUF(i);
        __syncthreads();   // everyone done reading the buffer we overwrite next
#if defined(HAVE_TDM)
        if (i + 1 < nIter) {
            const int k1 = (i + 1) * KC;
            const int nxt = BUF(i + 1);
            if (wid == 0) tdm_load_2d(lds_ofs(&xs[nxt][0]),  xbf + (size_t)m0 * DIM + k1, TILE_M, DIM, DIM - k1, TOKENS - m0);
            if (wid == 1) tdm_load_2d(lds_ofs(&w1s[nxt][0]), w1p + (size_t)n0 * DIM + k1, TILE_N, DIM, DIM - k1, HID_PAD - n0);
            if (wid == 2) tdm_load_2d(lds_ofs(&w3s[nxt][0]), w3p + (size_t)n0 * DIM + k1, TILE_N, DIM, DIM - k1, HID_PAD - n0);
            __builtin_amdgcn_s_wait_tensorcnt(1);   // tile i done (in-order TDM)
        } else {
            __builtin_amdgcn_s_wait_tensorcnt(0);
        }
#else
        {
            const int k0 = i * KC;
            copy_tile(&xs[0][0],  xbf + (size_t)m0 * DIM + k0, TILE_M, DIM, tid);
            copy_tile(&w1s[0][0], w1p + (size_t)n0 * DIM + k0, TILE_N, DIM, tid);
            copy_tile(&w3s[0][0], w3p + (size_t)n0 * DIM + k0, TILE_N, DIM, tid);
        }
#endif
        __syncthreads();   // tile i visible to all waves

        v16bf a[2], b1[2], b3[2];
#pragma unroll
        for (int mi = 0; mi < 2; ++mi)
            a[mi] = load_frag(&xs[cur][0], 32 * wm + 16 * mi + (lane & 15), lane);
#pragma unroll
        for (int ni = 0; ni < 2; ++ni) {
            b1[ni] = load_frag(&w1s[cur][0], 32 * wn + 16 * ni + (lane & 15), lane);
            b3[ni] = load_frag(&w3s[cur][0], 32 * wn + 16 * ni + (lane & 15), lane);
        }
#pragma unroll
        for (int mi = 0; mi < 2; ++mi)
#pragma unroll
            for (int ni = 0; ni < 2; ++ni) {
                acc1[mi][ni] = wmma_bf16(a[mi], b1[ni], acc1[mi][ni]);
                acc3[mi][ni] = wmma_bf16(a[mi], b3[ni], acc3[mi][ni]);
            }
    }

    // SwiGLU epilogue.  C/D layout: M = r + (lane>=16)*8, N = lane&15.
    const int rsel = (lane >> 4) << 3;
#pragma unroll
    for (int mi = 0; mi < 2; ++mi)
#pragma unroll
        for (int ni = 0; ni < 2; ++ni)
#pragma unroll
            for (int r = 0; r < 8; ++r) {
                float g  = acc1[mi][ni][r];
                float u  = acc3[mi][ni][r];
                float hv = (g / (1.f + __expf(-g))) * u;   // silu(g)*u
                int row = m0 + 32 * wm + 16 * mi + rsel + r;
                int c   = n0 + 32 * wn + 16 * ni + (lane & 15);
                hbuf[(size_t)row * HID_PAD + c] = f2bf(hv);
            }
}

// ---------------------------------------------------------------------------
// FFN pass 2: out += combine[:,e] * (h @ w2^T), K = HID_PAD (zero-padded).
// grid = (T/128, D/64).
// ---------------------------------------------------------------------------
__global__ __launch_bounds__(256) void moe_ffn2(const unsigned short* __restrict__ hbuf,
                                                const unsigned short* __restrict__ w2p,
                                                const float* __restrict__ combine,
                                                int e,
                                                float* __restrict__ out) {
    __shared__ unsigned short hs [2][TILE_M * KC];   // 2 x 8 KB
    __shared__ unsigned short w2s[2][TILE_N * KC];   // 2 x 4 KB

    const int tid  = threadIdx.x;
    const int lane = tid & 31;
    const int wid  = tid >> 5;
    const int wm   = wid & 3;
    const int wn   = wid >> 2;
    const int m0   = blockIdx.x * TILE_M;
    const int n0   = blockIdx.y * TILE_N;

    v8f zf = {0.f, 0.f, 0.f, 0.f, 0.f, 0.f, 0.f, 0.f};
    v8f acc[2][2] = {{zf, zf}, {zf, zf}};

    const int nIter = HID_PAD / KC;
#if defined(HAVE_TDM)
    if (wid == 0) tdm_load_2d(lds_ofs(&hs[0][0]),  hbuf + (size_t)m0 * HID_PAD, TILE_M, HID_PAD, HID_PAD, TOKENS - m0);
    if (wid == 1) tdm_load_2d(lds_ofs(&w2s[0][0]), w2p + (size_t)n0 * HID_PAD, TILE_N, HID_PAD, HID_PAD, DIM - n0);
#endif
    for (int i = 0; i < nIter; ++i) {
        const int cur = BUF(i);
        __syncthreads();
#if defined(HAVE_TDM)
        if (i + 1 < nIter) {
            const int k1 = (i + 1) * KC;
            const int nxt = BUF(i + 1);
            if (wid == 0) tdm_load_2d(lds_ofs(&hs[nxt][0]),  hbuf + (size_t)m0 * HID_PAD + k1, TILE_M, HID_PAD, HID_PAD - k1, TOKENS - m0);
            if (wid == 1) tdm_load_2d(lds_ofs(&w2s[nxt][0]), w2p + (size_t)n0 * HID_PAD + k1, TILE_N, HID_PAD, HID_PAD - k1, DIM - n0);
            __builtin_amdgcn_s_wait_tensorcnt(1);
        } else {
            __builtin_amdgcn_s_wait_tensorcnt(0);
        }
#else
        {
            const int k0 = i * KC;
            copy_tile(&hs[0][0],  hbuf + (size_t)m0 * HID_PAD + k0, TILE_M, HID_PAD, tid);
            copy_tile(&w2s[0][0], w2p + (size_t)n0 * HID_PAD + k0, TILE_N, HID_PAD, tid);
        }
#endif
        __syncthreads();

        v16bf a[2], b[2];
#pragma unroll
        for (int mi = 0; mi < 2; ++mi)
            a[mi] = load_frag(&hs[cur][0], 32 * wm + 16 * mi + (lane & 15), lane);
#pragma unroll
        for (int ni = 0; ni < 2; ++ni)
            b[ni] = load_frag(&w2s[cur][0], 32 * wn + 16 * ni + (lane & 15), lane);
#pragma unroll
        for (int mi = 0; mi < 2; ++mi)
#pragma unroll
            for (int ni = 0; ni < 2; ++ni)
                acc[mi][ni] = wmma_bf16(a[mi], b[ni], acc[mi][ni]);
    }

    const int rsel = (lane >> 4) << 3;
#pragma unroll
    for (int mi = 0; mi < 2; ++mi)
#pragma unroll
        for (int r = 0; r < 8; ++r) {
            int row = m0 + 32 * wm + 16 * mi + rsel + r;
            float scale = combine[(size_t)row * NUM_EXPERTS + e];
#pragma unroll
            for (int ni = 0; ni < 2; ++ni) {
                int c = n0 + 32 * wn + 16 * ni + (lane & 15);
                size_t o = (size_t)row * DIM + c;
                out[o] += scale * acc[mi][ni][r];   // experts serialized on stream
            }
        }
}

// ---------------------------------------------------------------------------
// Host launch.  ws layout (bytes):
//   combine  T*8*4        =   262144
//   aux      64
//   hbuf     T*704*2      = 11534336
//   xbf      T*1024*2     = 16777216
//   w1bf     8*704*1024*2 = 11534336
//   w3bf     8*704*1024*2 = 11534336
//   w2bf     8*1024*704*2 = 11534336   (total ~63.2 MB)
// ---------------------------------------------------------------------------
extern "C" void kernel_launch(void* const* d_in, const int* in_sizes, int n_in,
                              void* d_out, int out_size, void* d_ws, size_t ws_size,
                              hipStream_t stream) {
    (void)in_sizes; (void)n_in; (void)out_size; (void)ws_size;
    const float* x  = (const float*)d_in[0];
    const float* rw = (const float*)d_in[1];
    const float* w1 = (const float*)d_in[2];
    const float* w2 = (const float*)d_in[3];
    const float* w3 = (const float*)d_in[4];
    float* out = (float*)d_out;

    char* ws = (char*)d_ws;
    size_t off = 0;
    float* combine = (float*)(ws + off);            off += (size_t)TOKENS * NUM_EXPERTS * 4;
    float* aux     = (float*)(ws + off);            off += 64;
    unsigned short* hbuf = (unsigned short*)(ws + off); off += (size_t)TOKENS * HID_PAD * 2;
    unsigned short* xbf  = (unsigned short*)(ws + off); off += (size_t)TOKENS * DIM * 2;
    unsigned short* w1bf = (unsigned short*)(ws + off); off += (size_t)NUM_EXPERTS * HID_PAD * DIM * 2;
    unsigned short* w3bf = (unsigned short*)(ws + off); off += (size_t)NUM_EXPERTS * HID_PAD * DIM * 2;
    unsigned short* w2bf = (unsigned short*)(ws + off);

    moe_zero<<<(TOKENS * (size_t)DIM) / (256 * 4), 256, 0, stream>>>(out, aux);
    moe_router<<<TOKENS / 256, 256, 0, stream>>>(x, rw, combine, aux);
    moe_aux<<<1, 64, 0, stream>>>(aux, out + (size_t)TOKENS * DIM);

    // one-time bf16 conversion / padding
    conv_bf16_pad<<<dim3(TOKENS * (DIM / 4) / 256, 1), 256, 0, stream>>>(
        x, xbf, TOKENS, DIM, DIM, 0, 0);
    conv_bf16_pad<<<dim3(HID_PAD * (DIM / 4) / 256, NUM_EXPERTS), 256, 0, stream>>>(
        w1, w1bf, HID, DIM, DIM, (long long)HID * DIM, (long long)HID_PAD * DIM);
    conv_bf16_pad<<<dim3(HID_PAD * (DIM / 4) / 256, NUM_EXPERTS), 256, 0, stream>>>(
        w3, w3bf, HID, DIM, DIM, (long long)HID * DIM, (long long)HID_PAD * DIM);
    conv_bf16_pad<<<dim3(DIM * (HID_PAD / 4) / 256, NUM_EXPERTS), 256, 0, stream>>>(
        w2, w2bf, DIM, HID, HID_PAD, (long long)DIM * HID, (long long)DIM * HID_PAD);

    for (int e = 0; e < NUM_EXPERTS; ++e) {
        const unsigned short* w1p = w1bf + (size_t)e * HID_PAD * DIM;
        const unsigned short* w3p = w3bf + (size_t)e * HID_PAD * DIM;
        const unsigned short* w2p = w2bf + (size_t)e * DIM * HID_PAD;
        moe_ffn1<<<dim3(TOKENS / TILE_M, HID_PAD / TILE_N), 256, 0, stream>>>(
            xbf, w1p, w3p, hbuf);
        moe_ffn2<<<dim3(TOKENS / TILE_M, DIM / TILE_N), 256, 0, stream>>>(
            hbuf, w2p, combine, e, out);
    }
}
